// CaMoE_Block_18425409699867
// MI455X (gfx1250) — compile-verified, matrix operand order checked by reading, not run on hardware
//
#include <hip/hip_runtime.h>
#include <hip/hip_bf16.h>

// Problem dims
#define NTOK 8192          // B*T
#define CDIM 1024          // C
#define HDIM 4096          // H = 4*C
#define NEXP 5             // experts
#define HC   1024          // H chunk per LDS pass (4 chunks)

typedef __attribute__((ext_vector_type(16))) __bf16 v16bf;
typedef __attribute__((ext_vector_type(8)))  float  v8f;

// Hardware f32 -> bf16 conversion (RNE on CDNA5); single v_cvt instead of
// the 6-VALU manual round-to-nearest-even bit sequence.
__device__ __forceinline__ __bf16 f2bf(float f) {
  return (__bf16)f;
}

// K index of fragment element j for lane-half h0 (16-bit WMMA A/B layout)
// K = (j&7) + 8*h0 + 16*(j>>3)

// ---------------------------------------------------------------------------
// Kernel 0: pack a row-major fp32 [K, N] matrix into WMMA-B-fragment-major
// bf16 blocks. Block unit = (ntile, kkblk): 32 lanes x 16 bf16 contiguous,
// so the GEMM loads one coalesced 32B run per lane per WMMA.
// dst[((ntile*(K/32)+kkblk)*32 + lane)*16 + j] = bf16(src[k(j,h0)*N + ntile*16 + n15])
// ---------------------------------------------------------------------------
__global__ __launch_bounds__(256)
void pack_wmma_b(const float* __restrict__ src, __bf16* __restrict__ dst,
                 int K, int N) {
  const int unit = blockIdx.x * 8 + (threadIdx.x >> 5);  // (ntile * K/32 + kkblk)
  const int lane = threadIdx.x & 31;
  const int nblk = K >> 5;
  const int ntile = unit / nblk;
  const int kkblk = unit - ntile * nblk;
  if (ntile >= (N >> 4)) return;
  const int h0 = lane >> 4;
  const int n  = ntile * 16 + (lane & 15);
  const int kb = kkblk * 32 + 8 * h0;

  v16bf v;
#pragma unroll
  for (int j = 0; j < 16; ++j) {
    const int k = kb + (j & 7) + 16 * (j >> 3);
    reinterpret_cast<__bf16*>(&v)[j] = f2bf(src[(size_t)k * N + n]);
  }
  __bf16* d = dst + ((size_t)unit * 32 + lane) * 16;     // 32B aligned
  reinterpret_cast<uint4*>(d)[0] = reinterpret_cast<uint4*>(&v)[0];
  reinterpret_cast<uint4*>(d)[1] = reinterpret_cast<uint4*>(&v)[1];
}

// ---------------------------------------------------------------------------
// Kernel 1: LayerNorm -> h (bf16), winner confidence -> scale, expert buckets.
// One wave (32 lanes) per token; 8 waves per block.
// ---------------------------------------------------------------------------
__global__ __launch_bounds__(256)
void camoe_prep(const float* __restrict__ x,
                const int*   __restrict__ winners,
                const float* __restrict__ gamma,
                const float* __restrict__ beta,
                const float* __restrict__ wc,
                const float* __restrict__ bc,
                __bf16*      __restrict__ h_bf,
                float*       __restrict__ scale,
                int*         __restrict__ counts,
                int*         __restrict__ bucket) {
  const int wave = threadIdx.x >> 5;
  const int lane = threadIdx.x & 31;
  const int tok  = blockIdx.x * 8 + wave;
  if (tok >= NTOK) return;

  const float* xr = x + (size_t)tok * CDIM;
  float v[CDIM / 32];
  float sum = 0.f, ssq = 0.f;
#pragma unroll
  for (int i = 0; i < CDIM / 32; ++i) {
    float t = xr[lane + 32 * i];
    v[i] = t; sum += t; ssq += t * t;
  }
#pragma unroll
  for (int off = 16; off >= 1; off >>= 1) {
    sum += __shfl_xor(sum, off, 32);
    ssq += __shfl_xor(ssq, off, 32);
  }
  const float mean = sum * (1.0f / CDIM);
  const float var  = ssq * (1.0f / CDIM) - mean * mean;
  const float rstd = rsqrtf(var + 1e-5f);

  const int win = winners[tok];
  const float* wcr = wc + (size_t)win * CDIM;
  __bf16* hr = h_bf + (size_t)tok * CDIM;

  float dot = 0.f;
#pragma unroll
  for (int i = 0; i < CDIM / 32; ++i) {
    const int c = lane + 32 * i;
    float h = (v[i] - mean) * rstd * gamma[c] + beta[c];
    hr[c] = f2bf(h);
    dot += h * wcr[c];
  }
#pragma unroll
  for (int off = 16; off >= 1; off >>= 1) dot += __shfl_xor(dot, off, 32);

  if (lane == 0) {
    const float conf = 1.0f / (1.0f + expf(-(dot + bc[win])));
    scale[tok] = conf / (conf + 1e-6f);
    const int pos = atomicAdd(&counts[win], 1);
    bucket[win * NTOK + pos] = tok;
  }
}

// ---------------------------------------------------------------------------
// Kernel 2: grouped top-1 FFN:  out = x + scale * ( relu(h@W1[e])^2 @ W2[e] )
// Grid = NEXP * (NTOK/16) tile slots; block = 256 threads (8 waves).
// B fragments come pre-packed in bf16; one coalesced 32B load/lane per WMMA.
// ---------------------------------------------------------------------------
__global__ __launch_bounds__(256)
void camoe_ffn(const float*  __restrict__ x,
               const __bf16* __restrict__ w1p,    // packed [E][H/16][C/32][32][16]
               const __bf16* __restrict__ w2p,    // packed [E][C/16][H/32][32][16]
               const __bf16* __restrict__ h_bf,
               const float*  __restrict__ scale,
               const int*    __restrict__ counts,
               const int*    __restrict__ bucket,
               float*        __restrict__ out) {
  __shared__ __bf16 h_tile[16 * CDIM];   // 32 KB
  __shared__ __bf16 k_chunk[16 * HC];    // 32 KB

  const int e    = blockIdx.x / (NTOK / 16);
  const int tile = blockIdx.x % (NTOK / 16);
  const int cnt  = counts[e];
  if (tile * 16 >= cnt) return;
  const int nvalid = min(16, cnt - tile * 16);

  const int tid  = threadIdx.x;
  const int wv   = tid >> 5;
  const int lane = tid & 31;
  const int m    = lane & 15;
  const int h0   = lane >> 4;

  const int rsel   = (m < nvalid) ? m : 0;
  const int tokm   = bucket[e * NTOK + tile * 16 + rsel];
  const float sclm = scale[tokm];

  // Stage the 16 h rows into LDS (uint4 = 8 bf16 per load, coalesced).
#pragma unroll
  for (int it = 0; it < 8; ++it) {
    const int gi  = tid + it * 256;
    const int row = gi >> 7;
    const int col = gi & 127;
    const int trow = __shfl(tokm, row, 32);
    const uint4* src = reinterpret_cast<const uint4*>(h_bf + (size_t)trow * CDIM) + col;
    *reinterpret_cast<uint4*>(&h_tile[row * CDIM + col * 8]) = *src;
  }
  __syncthreads();

  const __bf16* w1e = w1p + (size_t)e * CDIM * HDIM;
  const __bf16* w2e = w2p + (size_t)e * HDIM * CDIM;

  v8f zero = {};
  v8f accC[8];
#pragma unroll
  for (int c = 0; c < 8; ++c) accC[c] = zero;

  for (int hb = 0; hb < HDIM; hb += HC) {
    // ---- stage 1: k_chunk = relu(h_tile @ W1[:, hb:hb+HC])^2  (M=16,N=HC,K=C)
#pragma unroll 1
    for (int j = 0; j < (HC / 16) / 8; ++j) {       // 8 col-tiles per wave
      const int ctl   = wv * 8 + j;                 // local col tile 0..63
      const int ntile = (hb >> 4) + ctl;            // global H col tile
      const __bf16* bbase =
          w1e + ((size_t)ntile * (CDIM / 32) * 32 + lane) * 16;
      v8f acc = zero;
#pragma unroll 4
      for (int kk = 0; kk < CDIM; kk += 32) {
        v16bf a;
        const __bf16* ar = &h_tile[m * CDIM + kk + 8 * h0];
        reinterpret_cast<uint4*>(&a)[0] = *reinterpret_cast<const uint4*>(ar);
        reinterpret_cast<uint4*>(&a)[1] = *reinterpret_cast<const uint4*>(ar + 16);
        const __bf16* bp = bbase + (size_t)(kk >> 5) * 32 * 16;
        v16bf b = *reinterpret_cast<const v16bf*>(bp);   // 32B coalesced
        __builtin_prefetch(bp + 32 * 16, 0, 1);
        acc = __builtin_amdgcn_wmma_f32_16x16x32_bf16(
            false, a, false, b, (short)0, acc, false, false);
      }
      // relu^2 -> bf16 -> LDS  (D layout: lane m = col, VGPR r -> row r+8*h0)
#pragma unroll
      for (int r = 0; r < 8; ++r) {
        const float t = fmaxf(acc[r], 0.0f);
        k_chunk[(r + 8 * h0) * HC + ctl * 16 + m] = f2bf(t * t);
      }
    }
    __syncthreads();

    // ---- stage 2: accC += k_chunk @ W2[hb:hb+HC, :]   (M=16,N=C,K=HC)
#pragma unroll 4
    for (int kk = 0; kk < HC; kk += 32) {
      v16bf a;
      const __bf16* ar = &k_chunk[m * HC + kk + 8 * h0];
      reinterpret_cast<uint4*>(&a)[0] = *reinterpret_cast<const uint4*>(ar);
      reinterpret_cast<uint4*>(&a)[1] = *reinterpret_cast<const uint4*>(ar + 16);
      const int kkblk = (hb + kk) >> 5;
#pragma unroll
      for (int c = 0; c < 8; ++c) {
        const int ntile = wv * 8 + c;               // output col tile in C
        const __bf16* bp =
            w2e + (((size_t)ntile * (HDIM / 32) + kkblk) * 32 + lane) * 16;
        v16bf b = *reinterpret_cast<const v16bf*>(bp);
        accC[c] = __builtin_amdgcn_wmma_f32_16x16x32_bf16(
            false, a, false, b, (short)0, accC[c], false, false);
      }
    }
    __syncthreads();
  }

  // ---- epilogue: out = x + scale * y (each wave owns a 16x128 slab)
#pragma unroll
  for (int r = 0; r < 8; ++r) {
    const int row = r + 8 * h0;
    const int trow = __shfl(tokm, row, 32);
    const float s  = __shfl(sclm, row, 32);
    if (row < nvalid) {
#pragma unroll
      for (int c = 0; c < 8; ++c) {
        const int ncol = (wv * 8 + c) * 16 + m;
        const size_t o = (size_t)trow * CDIM + ncol;
        out[o] = x[o] + accC[c][r] * s;
      }
    }
  }
}

// ---------------------------------------------------------------------------
extern "C" void kernel_launch(void* const* d_in, const int* in_sizes, int n_in,
                              void* d_out, int out_size, void* d_ws, size_t ws_size,
                              hipStream_t stream) {
  const float* x       = (const float*)d_in[0];
  const int*   winners = (const int*)  d_in[1];
  const float* gamma   = (const float*)d_in[2];
  const float* beta    = (const float*)d_in[3];
  const float* w1      = (const float*)d_in[4];
  const float* w2      = (const float*)d_in[5];
  const float* wc      = (const float*)d_in[6];
  const float* bc      = (const float*)d_in[7];
  float* out = (float*)d_out;

  // Workspace layout
  char* ws = (char*)d_ws;
  size_t off = 0;
  __bf16* h_bf  = (__bf16*)(ws + off); off += (size_t)NTOK * CDIM * 2;   // 16 MB
  float*  scale = (float*)(ws + off);  off += (size_t)NTOK * 4;
  int*    counts = (int*)(ws + off);   off += 64;
  int*    bucket = (int*)(ws + off);   off += (size_t)NEXP * NTOK * 4;
  off = (off + 255) & ~(size_t)255;
  __bf16* w1p = (__bf16*)(ws + off);   off += (size_t)NEXP * CDIM * HDIM * 2; // 42 MB
  __bf16* w2p = (__bf16*)(ws + off);   off += (size_t)NEXP * HDIM * CDIM * 2; // 42 MB

  hipMemsetAsync(counts, 0, 16 * sizeof(int), stream);

  // Pack weights into WMMA-B-fragment-major bf16 (per expert).
  for (int e = 0; e < NEXP; ++e) {
    // W1[e]: [K=C, N=H] -> units = (H/16)*(C/32) = 8192 waves = 1024 blocks
    pack_wmma_b<<<(HDIM / 16) * (CDIM / 32) / 8, 256, 0, stream>>>(
        w1 + (size_t)e * CDIM * HDIM, w1p + (size_t)e * CDIM * HDIM, CDIM, HDIM);
    // W2[e]: [K=H, N=C] -> units = (C/16)*(H/32) = 8192 waves = 1024 blocks
    pack_wmma_b<<<(CDIM / 16) * (HDIM / 32) / 8, 256, 0, stream>>>(
        w2 + (size_t)e * HDIM * CDIM, w2p + (size_t)e * HDIM * CDIM, HDIM, CDIM);
  }

  camoe_prep<<<NTOK / 8, 256, 0, stream>>>(x, winners, gamma, beta, wc, bc,
                                           h_bf, scale, counts, bucket);

  camoe_ffn<<<NEXP * (NTOK / 16), 256, 0, stream>>>(x, w1p, w2p, h_bf, scale,
                                                    counts, bucket, out);
}
